// Eigen2_2113123910052
// MI455X (gfx1250) — matrix operand order repaired and checked by reading, not compiled, not run on hardware
//
#include <hip/hip_runtime.h>
#include <hip/hip_bf16.h>
#include <math.h>

// ---------------- problem constants (match reference) ----------------
#define NN      2048      // nodes
#define EE      131072    // edges
#define KK      64        // eigenvector columns consumed (cols 1..64)
#define KP      80        // padded subspace width (>= KK+1, multiple of 16)
#define T_ITERS 16        // subspace-iteration rounds

typedef __attribute__((ext_vector_type(16))) __bf16 v16bf;
typedef __attribute__((ext_vector_type(8)))  float  v8f;

// ---------------- small utility kernels ----------------
__global__ void k_zero_f32(float* __restrict__ p, int n) {
    int i = blockIdx.x * blockDim.x + threadIdx.x;
    if (i < n) p[i] = 0.0f;
}

__global__ void k_scatter_adj(const int* __restrict__ ei, float* __restrict__ A) {
    int e = blockIdx.x * blockDim.x + threadIdx.x;
    if (e < EE) {
        int s = ei[e];
        int d = ei[EE + e];
        // counts are small integers -> exact in f32 -> order independent (deterministic)
        atomicAdd(&A[s * NN + d], 1.0f);
    }
}

__global__ void k_cvt_bf16(const float* __restrict__ A, __bf16* __restrict__ Ab, int n) {
    int i = blockIdx.x * blockDim.x + threadIdx.x;
    if (i < n) Ab[i] = (__bf16)A[i];
}

__global__ void k_init_V(float* __restrict__ Vf, __bf16* __restrict__ Vb) {
    int i = blockIdx.x * blockDim.x + threadIdx.x;
    if (i < NN * KP) {
        unsigned h = (unsigned)i * 2654435761u;
        h ^= h >> 15; h *= 2246822519u; h ^= h >> 13;
        float v = ((float)(h & 0xFFFFu) * (1.0f / 65536.0f)) - 0.5f;
        Vf[i] = v;
        Vb[i] = (__bf16)v;
    }
}

// ---------------- WMMA GEMM: W[NN x KP] = A[NN x NN] * V[NN x KP] ----------------
// 8 waves / workgroup; each wave owns a 16-row strip of a 128-row block and all
// 5 column tiles (KP = 5*16). K is staged through LDS in 32-wide chunks.
// A tile staged with 128-bit copies; V tile staged transposed so that both WMMA
// fragment gathers are contiguous (compiler emits ds_load_b128).
__global__ __launch_bounds__(256)
void k_gemm_AV(const __bf16* __restrict__ A, const __bf16* __restrict__ V,
               float* __restrict__ W) {
    __shared__ __bf16 As[128 * 32];     // A tile, row-major [r][k]
    __shared__ __bf16 Bs[KP * 32];      // V tile, TRANSPOSED: Bs[n*32 + k]

    const int tid   = threadIdx.x;
    const int wave  = tid >> 5;
    const int lane  = tid & 31;
    const int l16   = lane & 15;
    const int kh    = lane >> 4;        // 0: low K half, 1: high K half
    const int mBase = blockIdx.x * 128;
    const int waveM = wave * 16;        // row offset of this wave inside block

    // A-tile staging assignment: one aligned 32-byte chunk per thread
    const int stR  = tid >> 1;          // row 0..127
    const int stC  = (tid & 1) * 16;    // column half: 0 or 16

    v8f acc[5];
#pragma unroll
    for (int t = 0; t < 5; ++t) acc[t] = (v8f){0,0,0,0,0,0,0,0};

    for (int kb = 0; kb < NN; kb += 32) {
        // prefetch next A tile chunk (lowers to global_prefetch_b8)
        if (kb + 32 < NN)
            __builtin_prefetch(&A[(size_t)(mBase + stR) * NN + kb + 32 + stC], 0, 1);

        // stage A tile: 128x32 bf16, 32 contiguous bytes per thread (2 x b128)
        {
            const uint4* g = (const uint4*)(A + (size_t)(mBase + stR) * NN + kb + stC);
            uint4* l = (uint4*)(As + stR * 32 + stC);
            l[0] = g[0];
            l[1] = g[1];
        }
        // stage V tile transposed: Bs[n][k] = V[kb+k][n]
        // n fastest -> coalesced global reads; strided LDS scatter
        for (int idx = tid; idx < 32 * KP; idx += 256) {
            int k = idx / KP;
            int n = idx - k * KP;
            Bs[n * 32 + k] = V[(size_t)(kb + k) * KP + n];
        }
        __syncthreads();

        // A fragment (16x32 bf16): lanes 0-15 row M=l16 K=0..15,
        //                          lanes 16-31 row M=l16 K=16..31
        v16bf af;
#pragma unroll
        for (int i = 0; i < 8; ++i)
            af[i] = As[(waveM + l16) * 32 + kh * 16 + i];
#pragma unroll
        for (int i = 0; i < 8; ++i)
            af[8 + i] = As[(waveM + l16) * 32 + kh * 16 + 8 + i];

        // B fragments (32x16 bf16): lane holds column N=l16; lanes 0-15 K=0..15,
        // lanes 16-31 K=16..31. Bs transposed -> contiguous gather.
#pragma unroll
        for (int t = 0; t < 5; ++t) {
            v16bf bfrag;
#pragma unroll
            for (int i = 0; i < 16; ++i)
                bfrag[i] = Bs[(t * 16 + l16) * 32 + kh * 16 + i];
            acc[t] = __builtin_amdgcn_wmma_f32_16x16x32_bf16(
                /*neg_a=*/false, af, /*neg_b=*/false, bfrag,
                /*c_mod=*/(short)0, acc[t], /*reuse_a=*/false, /*reuse_b=*/false);
        }
        __syncthreads();
    }

    // C/D layout: VGPR r -> (M = kh*8 + r, N = l16)
#pragma unroll
    for (int t = 0; t < 5; ++t) {
#pragma unroll
        for (int r = 0; r < 8; ++r) {
            int row = mBase + waveM + kh * 8 + r;
            int col = t * 16 + l16;
            W[(size_t)row * KP + col] = acc[t][r];
        }
    }
}

// ---------------- Gram matrix: G = W^T W  (KP x KP) ----------------
__global__ __launch_bounds__(256)
void k_gram(const float* __restrict__ W, float* __restrict__ G) {
    int i = blockIdx.x / KP;
    int j = blockIdx.x % KP;
    __shared__ float red[256];
    float s = 0.0f;
    for (int r = threadIdx.x; r < NN; r += 256)
        s += W[r * KP + i] * W[r * KP + j];
    red[threadIdx.x] = s;
    __syncthreads();
    for (int off = 128; off > 0; off >>= 1) {
        if (threadIdx.x < off) red[threadIdx.x] += red[threadIdx.x + off];
        __syncthreads();
    }
    if (threadIdx.x == 0) G[i * KP + j] = red[0];
}

// ---------------- Cholesky of G (regularized) + lower-tri inverse ----------------
__global__ __launch_bounds__(128)
void k_chol_inv(const float* __restrict__ G, float* __restrict__ Linv) {
    __shared__ float L[KP * KP];
    const int tid = threadIdx.x;
    for (int idx = tid; idx < KP * KP; idx += 128) {
        int i = idx / KP, j = idx % KP;
        float g = G[idx];
        if (i == j) g += 1e-5f;          // regularize
        L[idx] = (j <= i) ? g : 0.0f;
    }
    __syncthreads();
    for (int k = 0; k < KP; ++k) {
        if (tid == 0) L[k * KP + k] = sqrtf(fmaxf(L[k * KP + k], 1e-20f));
        __syncthreads();
        float dk = L[k * KP + k];
        for (int i = k + 1 + tid; i < KP; i += 128) L[i * KP + k] /= dk;
        __syncthreads();
        for (int i = k + 1 + tid; i < KP; i += 128) {
            float lik = L[i * KP + k];
            for (int j = k + 1; j <= i; ++j)
                L[i * KP + j] -= lik * L[j * KP + k];
        }
        __syncthreads();
    }
    // forward substitution per column: Linv = L^-1 (lower triangular)
    if (tid < KP) {
        int c = tid;
        for (int i = 0; i < KP; ++i) Linv[i * KP + c] = 0.0f;
        for (int i = c; i < KP; ++i) {
            float s = (i == c) ? 1.0f : 0.0f;
            for (int j = c; j < i; ++j) s -= L[i * KP + j] * Linv[j * KP + c];
            Linv[i * KP + c] = s / L[i * KP + i];
        }
    }
}

// ---------------- V = W * Linv^T   (orthonormalized basis, f32 + bf16 copies) ----
__global__ void k_apply_R(const float* __restrict__ W, const float* __restrict__ Linv,
                          float* __restrict__ Vf, __bf16* __restrict__ Vb) {
    int idx = blockIdx.x * blockDim.x + threadIdx.x;
    if (idx >= NN * KP) return;
    int r = idx / KP, j = idx % KP;
    float s = 0.0f;
    for (int c = 0; c <= j; ++c)          // Linv is lower triangular
        s += W[r * KP + c] * Linv[j * KP + c];
    Vf[idx] = s;
    Vb[idx] = (__bf16)s;
}

// ---------------- Rayleigh quotients: lam[j] = V[:,j] . (A V)[:,j] --------------
__global__ __launch_bounds__(256)
void k_rayleigh(const float* __restrict__ Vf, const float* __restrict__ W,
                float* __restrict__ lam) {
    int j = blockIdx.x;
    __shared__ float red[256];
    float s = 0.0f;
    for (int r = threadIdx.x; r < NN; r += 256)
        s += Vf[r * KP + j] * W[r * KP + j];
    red[threadIdx.x] = s;
    __syncthreads();
    for (int off = 128; off > 0; off >>= 1) {
        if (threadIdx.x < off) red[threadIdx.x] += red[threadIdx.x + off];
        __syncthreads();
    }
    if (threadIdx.x == 0) lam[j] = red[0];
}

// ---------------- output packing: eigenvals (complex-interleaved) + top_eig ------
__global__ void k_write_eigs(const float* __restrict__ lam, const float* __restrict__ Vf,
                             float* __restrict__ out, int out_size) {
    int i = blockIdx.x * blockDim.x + threadIdx.x;
    // eigenvals region: out[EE .. EE + 2*NN) as (re, im) pairs, im = 0
    if (i < 2 * NN) {
        int o = EE + i;
        float v = 0.0f;
        if ((i & 1) == 0) {
            int j = i >> 1;
            if (j < KP) v = lam[j];
        }
        if (o < out_size) out[o] = v;
    }
    // top_eig region: columns 1..KK of V -> out[EE + 2*NN ...]
    const int base = EE + 2 * NN;
    for (int idx = i; idx < NN * KK; idx += gridDim.x * blockDim.x) {
        int r = idx / KK, j = idx % KK;
        int o = base + idx;
        if (o < out_size) out[o] = Vf[r * KP + 1 + j];
    }
}

// ---------------- edge features: 8*sqrt(ln n)*sigma * sign(<v_i, v_j>) ----------
__global__ void k_edge_feat(const int* __restrict__ ei, const float* __restrict__ Vf,
                            const float* __restrict__ sigma, float* __restrict__ out,
                            int out_size) {
    int e = blockIdx.x * blockDim.x + threadIdx.x;
    if (e >= EE) return;
    int s = ei[e];
    int d = ei[EE + e];
    float dot = 0.0f;
#pragma unroll 8
    for (int j = 0; j < KK; ++j)
        dot += Vf[s * KP + 1 + j] * Vf[d * KP + 1 + j];
    float sg = (dot > 0.0f) ? 1.0f : ((dot < 0.0f) ? -1.0f : 0.0f);
    float scale = 8.0f * sqrtf(logf((float)NN)) * sigma[0];
    if (e < out_size) out[e] = scale * sg;
}

// ---------------- orchestration ----------------
static inline size_t align_up(size_t x) { return (x + 255) & ~(size_t)255; }

extern "C" void kernel_launch(void* const* d_in, const int* in_sizes, int n_in,
                              void* d_out, int out_size, void* d_ws, size_t ws_size,
                              hipStream_t stream) {
    (void)in_sizes; (void)n_in; (void)ws_size;
    const int*   ei    = (const int*)d_in[0];    // [2, EE] flat
    const float* sigma = (const float*)d_in[2];  // [1]
    float*       out   = (float*)d_out;

    char* ws = (char*)d_ws;
    size_t off = 0;
    float*  A    = (float*) (ws + off); off = align_up(off + (size_t)NN * NN * 4);
    __bf16* Ab   = (__bf16*)(ws + off); off = align_up(off + (size_t)NN * NN * 2);
    float*  W    = (float*) (ws + off); off = align_up(off + (size_t)NN * KP * 4);
    float*  Vf   = (float*) (ws + off); off = align_up(off + (size_t)NN * KP * 4);
    __bf16* Vb   = (__bf16*)(ws + off); off = align_up(off + (size_t)NN * KP * 2);
    float*  G    = (float*) (ws + off); off = align_up(off + (size_t)KP * KP * 4);
    float*  Linv = (float*) (ws + off); off = align_up(off + (size_t)KP * KP * 4);
    float*  lam  = (float*) (ws + off); off = align_up(off + (size_t)KP * 4);

    // 1) dense adjacency (f32 exact counts) -> bf16
    k_zero_f32 <<<(NN * NN + 255) / 256, 256, 0, stream>>>(A, NN * NN);
    k_scatter_adj<<<(EE + 255) / 256, 256, 0, stream>>>(ei, A);
    k_cvt_bf16 <<<(NN * NN + 255) / 256, 256, 0, stream>>>(A, Ab, NN * NN);

    // 2) subspace iteration:  V <- qr(A V)  x T_ITERS, all heavy work in WMMA GEMM
    k_init_V<<<(NN * KP + 255) / 256, 256, 0, stream>>>(Vf, Vb);
    for (int it = 0; it < T_ITERS; ++it) {
        k_gemm_AV <<<NN / 128, 256, 0, stream>>>(Ab, Vb, W);
        k_gram    <<<KP * KP,  256, 0, stream>>>(W, G);
        k_chol_inv<<<1,        128, 0, stream>>>(G, Linv);
        k_apply_R <<<(NN * KP + 255) / 256, 256, 0, stream>>>(W, Linv, Vf, Vb);
    }

    // 3) eigenvalue estimates (Rayleigh quotients on final basis)
    k_gemm_AV <<<NN / 128, 256, 0, stream>>>(Ab, Vb, W);
    k_rayleigh<<<KP, 256, 0, stream>>>(Vf, W, lam);

    // 4) outputs: [features EE] [eigenvals 2*NN interleaved] [top_eig NN*KK]
    k_write_eigs<<<(NN * KK + 255) / 256, 256, 0, stream>>>(lam, Vf, out, out_size);
    k_edge_feat <<<(EE + 255) / 256, 256, 0, stream>>>(ei, Vf, sigma, out, out_size);
}
